// HEncoder_40596030882242
// MI455X (gfx1250) — matrix operand order repaired and checked by reading, not compiled, not run on hardware
//
#include <hip/hip_runtime.h>
#include <hip/hip_bf16.h>
#include <math.h>

typedef __attribute__((ext_vector_type(16))) _Float16 v16h;
typedef __attribute__((ext_vector_type(8)))  _Float16 v8h;
typedef __attribute__((ext_vector_type(8)))  float    v8f;

#define ACT_NONE 0
#define ACT_RELU_BN 1
#define ACT_GELU 2

// ---------------------------------------------------------------- WMMA helpers
__device__ __forceinline__ v8f wmma16x16x32(v16h a, v16h b, v8f c) {
  return __builtin_amdgcn_wmma_f32_16x16x32_f16(false, a, false, b, (short)0, c,
                                                false, false);
}

// A fragment (16xK tile slice, K chunk of 32), row-major f16, lda in halfs.
// lane 0-15: row=lane, elems 0-15 = K k0..k0+15
// lane16-31: row=lane-16, elems 0-15 = K k0+16..k0+31
__device__ __forceinline__ v16h load_a_frag(const _Float16* A, int lda, int k0,
                                            int lane) {
  int r = lane & 15, hf = lane >> 4;
  const _Float16* p = A + (size_t)r * lda + k0 + hf * 16;
  v8h lo = *(const v8h*)(p);
  v8h hi = *(const v8h*)(p + 8);
  v16h o;
#pragma unroll
  for (int i = 0; i < 8; ++i) { o[i] = lo[i]; o[i + 8] = hi[i]; }
  return o;
}

// A fragment with zero padding past kmax (kmax multiple of 8)
__device__ __forceinline__ v16h load_a_frag_pad(const _Float16* A, int lda,
                                                int k0, int kmax, int lane) {
  int r = lane & 15, hf = lane >> 4;
  int c0 = k0 + hf * 16;
  const _Float16* p = A + (size_t)r * lda;
  v8h z = {};
  v8h lo = (c0 + 8 <= kmax) ? *(const v8h*)(p + c0) : z;
  v8h hi = (c0 + 16 <= kmax) ? *(const v8h*)(p + c0 + 8) : z;
  v16h o;
#pragma unroll
  for (int i = 0; i < 8; ++i) { o[i] = lo[i]; o[i + 8] = hi[i]; }
  return o;
}

// B fragment from pre-transposed weights Wt[N][Kp] (row-major per output col).
// lane 0-15: col=lane, elems = K k0..k0+15; lane16-31: col=lane-16, K k0+16..31
__device__ __forceinline__ v16h load_b_frag(const _Float16* Wt, int ldb, int k0,
                                            int lane) {
  int c = lane & 15, hf = lane >> 4;
  return *(const v16h*)(Wt + (size_t)c * ldb + k0 + hf * 16);
}

// B fragment where B = M^T and M is row-major [col][k] with zero pad past kmax
// (kmax multiple of 16)
__device__ __forceinline__ v16h load_bT_frag_pad(const _Float16* Mrows, int ld,
                                                 int k0, int kmax, int lane) {
  int c = lane & 15, hf = lane >> 4;
  int koff = k0 + hf * 16;
  if (koff + 16 <= kmax) return *(const v16h*)(Mrows + (size_t)c * ld + koff);
  v16h z = {};
  return z;
}

// ---------------------------------------------------------------- small utils
__global__ void k_cast_pad(const float* __restrict__ X, _Float16* __restrict__ Y,
                           int M, int K, int KP) {
  int i = blockIdx.x * blockDim.x + threadIdx.x;
  if (i >= M * KP) return;
  int r = i / KP, c = i - r * KP;
  Y[i] = (c < K) ? (_Float16)X[(size_t)r * K + c] : (_Float16)0.f;
}

__global__ void k_gather_cast(const float* __restrict__ X,
                              const int* __restrict__ idx,
                              _Float16* __restrict__ Y, int R, int d) {
  int i = blockIdx.x * blockDim.x + threadIdx.x;
  if (i >= R * d) return;
  int r = i / d, c = i - r * d;
  Y[i] = (_Float16)X[(size_t)idx[r] * d + c];
}

// W[K][N] f32 -> Wt[N][KP] f16 (transposed, K zero padded to KP)
__global__ void k_prep_wt(const float* __restrict__ W, _Float16* __restrict__ Wt,
                          int K, int N, int KP) {
  int i = blockIdx.x * blockDim.x + threadIdx.x;
  if (i >= N * KP) return;
  int nr = i / KP, k = i - nr * KP;
  Wt[i] = (k < K) ? (_Float16)W[(size_t)k * N + nr] : (_Float16)0.f;
}

__global__ void k_add(const float* __restrict__ a, const float* __restrict__ p,
                      float* __restrict__ z, int nelem) {
  int i = blockIdx.x * blockDim.x + threadIdx.x;
  if (i < nelem) z[i] = a[i] + p[i];
}

// one wave32 per row: Y = f16(LN(X)*g+b)
__global__ void k_layernorm(const float* __restrict__ X,
                            const float* __restrict__ g,
                            const float* __restrict__ b,
                            _Float16* __restrict__ Y, int M, int d) {
  int row = blockIdx.x * (blockDim.x >> 5) + (threadIdx.x >> 5);
  if (row >= M) return;
  int lane = threadIdx.x & 31;
  const float* x = X + (size_t)row * d;
  float s = 0.f, s2 = 0.f;
  for (int c = lane; c < d; c += 32) { float v = x[c]; s += v; s2 += v * v; }
#pragma unroll
  for (int o = 16; o; o >>= 1) {
    s += __shfl_xor(s, o, 32);
    s2 += __shfl_xor(s2, o, 32);
  }
  float mean = s / d;
  float inv = rsqrtf(s2 / d - mean * mean + 1e-5f);
  for (int c = lane; c < d; c += 32)
    Y[(size_t)row * d + c] = (_Float16)((x[c] - mean) * inv * g[c] + b[c]);
}

// final LN fused with [B,n,d] -> [B,d,n] transpose into d_out (f32)
__global__ void k_ln_out(const float* __restrict__ X, const float* __restrict__ g,
                         const float* __restrict__ b, float* __restrict__ out,
                         int n, int d, int M) {
  int row = blockIdx.x * (blockDim.x >> 5) + (threadIdx.x >> 5);
  if (row >= M) return;
  int lane = threadIdx.x & 31;
  int bb = row / n, gi = row - bb * n;
  const float* x = X + (size_t)row * d;
  float s = 0.f, s2 = 0.f;
  for (int c = lane; c < d; c += 32) { float v = x[c]; s += v; s2 += v * v; }
#pragma unroll
  for (int o = 16; o; o >>= 1) {
    s += __shfl_xor(s, o, 32);
    s2 += __shfl_xor(s2, o, 32);
  }
  float mean = s / d;
  float inv = rsqrtf(s2 / d - mean * mean + 1e-5f);
  for (int c = lane; c < d; c += 32)
    out[((size_t)bb * d + c) * n + gi] = (x[c] - mean) * inv * g[c] + b[c];
}

// ------------------------------------------------------------- generic WMMA GEMM
// C[M,N] = epilogue(A[M,Kp] @ Wt[N,Kp]^T + bias).
// One wave per 32x32 macro-tile (2x2 register blocking of 16x16 WMMA tiles):
// per K-step, 2 A-frags + 2 B-frags feed 4 independent v_wmma chains
// (2 b128 loads per WMMA, 4 accumulators for co-issue). M,N multiples of 32.
__global__ void k_gemm(const _Float16* __restrict__ A, int lda,
                       const _Float16* __restrict__ Wt, int ldb,
                       const float* __restrict__ bias,
                       const float* __restrict__ bns,
                       const float* __restrict__ bnb,
                       const float* __restrict__ res, float* __restrict__ Cf,
                       _Float16* __restrict__ Ch, int ldch, int M, int N, int KP,
                       int act) {
  int lane = threadIdx.x & 31;
  int wv = threadIdx.x >> 5;
  int n2 = N >> 5;
  int m2 = M >> 5;
  int tile = blockIdx.x * (blockDim.x >> 5) + wv;
  if (tile >= m2 * n2) return;
  int mt = tile / n2, nt = tile - mt * n2;
  const _Float16* A0 = A + (size_t)(mt * 32) * lda;
  const _Float16* A1 = A0 + (size_t)16 * lda;
  const _Float16* W0 = Wt + (size_t)(nt * 32) * ldb;
  const _Float16* W1 = W0 + (size_t)16 * ldb;
  v8f acc[2][2] = {};
  for (int k0 = 0; k0 < KP; k0 += 32) {
    if (k0 + 32 < KP) {
      __builtin_prefetch(W0 + (size_t)(lane & 15) * ldb + k0 + 32, 0, 1);
      __builtin_prefetch(W1 + (size_t)(lane & 15) * ldb + k0 + 32, 0, 1);
    }
    v16h a0 = load_a_frag(A0, lda, k0, lane);
    v16h a1 = load_a_frag(A1, lda, k0, lane);
    v16h b0 = load_b_frag(W0, ldb, k0, lane);
    v16h b1 = load_b_frag(W1, ldb, k0, lane);
    acc[0][0] = wmma16x16x32(a0, b0, acc[0][0]);
    acc[0][1] = wmma16x16x32(a0, b1, acc[0][1]);
    acc[1][0] = wmma16x16x32(a1, b0, acc[1][0]);
    acc[1][1] = wmma16x16x32(a1, b1, acc[1][1]);
  }
#pragma unroll
  for (int mi = 0; mi < 2; ++mi) {
#pragma unroll
    for (int ni = 0; ni < 2; ++ni) {
      int col = nt * 32 + ni * 16 + (lane & 15);
      int rbase = mt * 32 + mi * 16 + ((lane >> 4) << 3);
      float bi = bias ? bias[col] : 0.f;
      float sc = bns ? bns[col] : 1.f;
      float sh = bnb ? bnb[col] : 0.f;
#pragma unroll
      for (int j = 0; j < 8; ++j) {
        int row = rbase + j;
        float v = acc[mi][ni][j] + bi;
        if (act == ACT_RELU_BN) {
          v = v * sc + sh;
          v = v > 0.f ? v : 0.f;
        } else if (act == ACT_GELU) {
          v = 0.5f * v * (1.f + erff(v * 0.70710678118f));
        }
        if (res) v += res[(size_t)row * N + col];
        if (Cf) Cf[(size_t)row * N + col] = v;
        if (Ch) Ch[(size_t)row * ldch + col] = (_Float16)v;
      }
    }
  }
}

// --------------------------------------------------- fused token-embed (per group)
// f1=relu((A@W1+b1)*s1+t1); f2=f1@W2+b2; fg=max_n f2; f3=relu((cat[fg,f2]@W3+b3)*s2+t2)
// tok = max_n (f3@W4+b4). All intermediates LDS-resident.
__global__ void k_token_embed(
    const _Float16* __restrict__ Ain, const _Float16* __restrict__ Wt1,
    const float* __restrict__ b1, const float* __restrict__ bn1s,
    const float* __restrict__ bn1b, const _Float16* __restrict__ Wt2,
    const float* __restrict__ b2, const _Float16* __restrict__ Wt3,
    const float* __restrict__ b3, const float* __restrict__ bn2s,
    const float* __restrict__ bn2b, const _Float16* __restrict__ Wt4,
    const float* __restrict__ b4, float* __restrict__ tok, int np, int icp,
    int hh, int hhp, int d) {
  extern __shared__ char sm[];
  _Float16* sF1 = (_Float16*)sm;            // np*hhp
  _Float16* sF2 = sF1 + np * hhp;           // np*hh
  _Float16* sF3 = sF2 + np * hh;            // np*d
  _Float16* sF3o = sF3 + np * d;            // np*d
  float* sRed = (float*)(sF3o + np * d);    // (np/16)*d
  const int lane = threadIdx.x & 31;
  const int wv = threadIdx.x >> 5;
  const int nw = blockDim.x >> 5;
  const int tid = threadIdx.x;
  const int mtiles = np >> 4;
  const _Float16* Arow = Ain + (size_t)blockIdx.x * np * icp;

  const int n1t = hh >> 4;
  for (int t = wv; t < mtiles * n1t; t += nw) {
    int mt = t / n1t, nt = t - mt * n1t;
    v8f acc = {};
    for (int k0 = 0; k0 < icp; k0 += 32)
      acc = wmma16x16x32(load_a_frag(Arow + (size_t)mt * 16 * icp, icp, k0, lane),
                         load_b_frag(Wt1 + (size_t)nt * 16 * icp, icp, k0, lane),
                         acc);
    int col = nt * 16 + (lane & 15);
    float bi = b1[col], sc = bn1s[col], sh = bn1b[col];
#pragma unroll
    for (int j = 0; j < 8; ++j) {
      int row = mt * 16 + ((lane >> 4) << 3) + j;
      float v = (acc[j] + bi) * sc + sh;
      sF1[row * hhp + col] = (_Float16)(v > 0.f ? v : 0.f);
    }
  }
  if (hhp > hh) {  // zero the K padding columns of f1
    int padw = hhp - hh;
    for (int i = tid; i < np * padw; i += blockDim.x) {
      int r = i / padw, c = hh + (i - r * padw);
      sF1[r * hhp + c] = (_Float16)0.f;
    }
  }
  __syncthreads();
  for (int t = wv; t < mtiles * n1t; t += nw) {
    int mt = t / n1t, nt = t - mt * n1t;
    v8f acc = {};
    for (int k0 = 0; k0 < hhp; k0 += 32)
      acc = wmma16x16x32(load_a_frag(sF1 + mt * 16 * hhp, hhp, k0, lane),
                         load_b_frag(Wt2 + (size_t)nt * 16 * hhp, hhp, k0, lane),
                         acc);
    int col = nt * 16 + (lane & 15);
    float bi = b2[col];
#pragma unroll
    for (int j = 0; j < 8; ++j) {
      int row = mt * 16 + ((lane >> 4) << 3) + j;
      sF2[row * hh + col] = (_Float16)(acc[j] + bi);
    }
  }
  __syncthreads();
  for (int c = tid; c < hh; c += blockDim.x) {  // global-feature max + concat
    float m = -1e30f;
    for (int r = 0; r < np; ++r) m = fmaxf(m, (float)sF2[r * hh + c]);
    _Float16 mh = (_Float16)m;
    for (int r = 0; r < np; ++r) {
      sF3[r * d + c] = mh;
      sF3[r * d + hh + c] = sF2[r * hh + c];
    }
  }
  __syncthreads();
  const int n3t = d >> 4;
  for (int t = wv; t < mtiles * n3t; t += nw) {
    int mt = t / n3t, nt = t - mt * n3t;
    v8f acc = {};
    for (int k0 = 0; k0 < d; k0 += 32)
      acc = wmma16x16x32(load_a_frag(sF3 + mt * 16 * d, d, k0, lane),
                         load_b_frag(Wt3 + (size_t)nt * 16 * d, d, k0, lane), acc);
    int col = nt * 16 + (lane & 15);
    float bi = b3[col], sc = bn2s[col], sh = bn2b[col];
#pragma unroll
    for (int j = 0; j < 8; ++j) {
      int row = mt * 16 + ((lane >> 4) << 3) + j;
      float v = (acc[j] + bi) * sc + sh;
      sF3o[row * d + col] = (_Float16)(v > 0.f ? v : 0.f);
    }
  }
  __syncthreads();
  for (int t = wv; t < mtiles * n3t; t += nw) {
    int mt = t / n3t, nt = t - mt * n3t;
    v8f acc = {};
    for (int k0 = 0; k0 < d; k0 += 32)
      acc = wmma16x16x32(load_a_frag(sF3o + mt * 16 * d, d, k0, lane),
                         load_b_frag(Wt4 + (size_t)nt * 16 * d, d, k0, lane), acc);
    int col = nt * 16 + (lane & 15);
    float bi = b4[col];
    float m = -1e30f;
#pragma unroll
    for (int j = 0; j < 8; ++j) m = fmaxf(m, acc[j] + bi);
    m = fmaxf(m, __shfl_xor(m, 16, 32));  // combine row halves (same column)
    if (lane < 16) sRed[mt * d + col] = m;
  }
  __syncthreads();
  for (int c = tid; c < d; c += blockDim.x) {
    float m = -1e30f;
    for (int mt = 0; mt < mtiles; ++mt) m = fmaxf(m, sRed[mt * d + c]);
    tok[(size_t)blockIdx.x * d + c] = m;
  }
}

// ----------------------------------------- radius-masked attention, LDS scores
// one workgroup per (batch, head, 16-query tile); S, P and V^T fully in LDS.
__global__ void k_attn(const _Float16* __restrict__ qkv,  // [B*n][3d]
                       const float* __restrict__ cen,     // [B*n][3]
                       _Float16* __restrict__ Oh,         // [B*n][d]
                       int n, int d, int hd, float scale, float r2) {
  extern __shared__ char sm2[];
  float* S = (float*)sm2;                               // [16][n]
  _Float16* Pm = (_Float16*)(sm2 + (size_t)16 * n * 4); // [16][n]
  _Float16* Vt = Pm + (size_t)16 * n;                   // [hd][n]
  const int nqt = n >> 4;
  int bid = blockIdx.x;
  int qt = bid % nqt;
  int h = (bid / nqt) % 6;
  int b = bid / (nqt * 6);
  const int lane = threadIdx.x & 31;
  const int wv = threadIdx.x >> 5;
  const int nw = blockDim.x >> 5;
  const int tid = threadIdx.x;
  const int ld = 3 * d;
  const int q0 = qt * 16;
  const int kpad = (hd + 31) & ~31;
  // phase 0: stage V^T in LDS (coalesced reads of the V slice of qkv)
  {
    const _Float16* Vg = qkv + (size_t)(b * n) * ld + 2 * d + h * hd;
    for (int i = tid; i < n * hd; i += blockDim.x) {
      int key = i / hd, c = i - key * hd;
      Vt[c * n + key] = Vg[(size_t)key * ld + c];
    }
  }
  const _Float16* Qb = qkv + (size_t)(b * n + q0) * ld + h * hd;
  // phase 1: S = scale*Q@K^T with on-the-fly radius mask
  for (int kt = wv; kt < nqt; kt += nw) {
    const _Float16* Kb = qkv + (size_t)(b * n + kt * 16) * ld + d + h * hd;
    v8f acc = {};
    for (int k0 = 0; k0 < kpad; k0 += 32)
      acc = wmma16x16x32(load_a_frag_pad(Qb, ld, k0, hd, lane),
                         load_bT_frag_pad(Kb, ld, k0, hd, lane), acc);
    int key = kt * 16 + (lane & 15);
    float kx = cen[(size_t)(b * n + key) * 3 + 0];
    float ky = cen[(size_t)(b * n + key) * 3 + 1];
    float kz = cen[(size_t)(b * n + key) * 3 + 2];
#pragma unroll
    for (int j = 0; j < 8; ++j) {
      int ql = ((lane >> 4) << 3) + j;
      int q = q0 + ql;
      float dx = cen[(size_t)(b * n + q) * 3 + 0] - kx;
      float dy = cen[(size_t)(b * n + q) * 3 + 1] - ky;
      float dz = cen[(size_t)(b * n + q) * 3 + 2] - kz;
      float v = acc[j] * scale;
      if (dx * dx + dy * dy + dz * dz >= r2) v = -1e9f;
      S[ql * n + key] = v;
    }
  }
  __syncthreads();
  // phase 2: row softmax (wave32 shuffle reductions)
  for (int r = wv; r < 16; r += nw) {
    float m = -1e30f;
    for (int c = lane; c < n; c += 32) m = fmaxf(m, S[r * n + c]);
#pragma unroll
    for (int o = 16; o; o >>= 1) m = fmaxf(m, __shfl_xor(m, o, 32));
    float ssum = 0.f;
    for (int c = lane; c < n; c += 32) ssum += __expf(S[r * n + c] - m);
#pragma unroll
    for (int o = 16; o; o >>= 1) ssum += __shfl_xor(ssum, o, 32);
    float inv = 1.f / ssum;
    for (int c = lane; c < n; c += 32)
      Pm[r * n + c] = (_Float16)(__expf(S[r * n + c] - m) * inv);
  }
  __syncthreads();
  // phase 3: O = P @ V ; B frags are contiguous 32B LDS reads from V^T
  for (int ot = wv; ot < (hd >> 4); ot += nw) {
    int vcol = ot * 16 + (lane & 15);
    v8f acc = {};
    for (int k0 = 0; k0 < n; k0 += 32) {
      v16h a = load_a_frag(Pm, n, k0, lane);
      int kb = k0 + ((lane >> 4) << 4);
      v16h bb = *(const v16h*)(Vt + (size_t)vcol * n + kb);
      acc = wmma16x16x32(a, bb, acc);
    }
#pragma unroll
    for (int j = 0; j < 8; ++j) {
      int ql = ((lane >> 4) << 3) + j;
      Oh[(size_t)(b * n + q0 + ql) * d + h * hd + ot * 16 + (lane & 15)] =
          (_Float16)acc[j];
    }
  }
}

// ------------------------------------------------------------------- host side
extern "C" void kernel_launch(void* const* d_in, const int* in_sizes, int n_in,
                              void* d_out, int out_size, void* d_ws,
                              size_t ws_size, hipStream_t stream) {
  (void)in_sizes; (void)n_in; (void)out_size; (void)ws_size;
  const int Bv = 16, Hn = 6;
  const int Gv[3] = {1024, 256, 64};
  const int Kv[3] = {32, 16, 16};
  const int Dv[3] = {96, 192, 384};
  const float RAD[3] = {0.32f, 0.64f, 1.28f};

  const float* neigh0 = (const float*)d_in[0];
  const float* cent[3] = {(const float*)d_in[1], (const float*)d_in[2],
                          (const float*)d_in[3]};
  const int* idxs[2] = {(const int*)d_in[4], (const int*)d_in[5]};
  auto P = [&](int i) { return (const float*)d_in[6 + i]; };

  size_t off = 0;
  auto alloc = [&](size_t bytes) -> void* {
    off = (off + 255) & ~(size_t)255;
    void* p = (char*)d_ws + off;
    off += bytes;
    return p;
  };
  auto prep = [&](const float* W, int K, int N, _Float16* Wt, int KP) {
    int tot = N * KP;
    k_prep_wt<<<(tot + 255) / 256, 256, 0, stream>>>(W, Wt, K, N, KP);
  };
  auto gemm = [&](const _Float16* A, int lda, const _Float16* Wt, int ldb,
                  const float* bias, const float* bns, const float* bnb,
                  const float* res, float* Cf, _Float16* Ch, int ldch, int M,
                  int N, int KP, int act) {
    int tiles = (M / 32) * (N / 32);
    k_gemm<<<(tiles + 3) / 4, 128, 0, stream>>>(A, lda, Wt, ldb, bias, bns, bnb,
                                                res, Cf, Ch, ldch, M, N, KP, act);
  };

  float* outp = (float*)d_out;
  size_t outoff[3] = {0, (size_t)16 * 96 * 1024,
                      (size_t)16 * 96 * 1024 + (size_t)16 * 192 * 256};

  const float* xprev = nullptr;
  int dprev = 0;
  for (int s = 0; s < 3; ++s) {
    const int n = Gv[s], d = Dv[s], hh = d / 2, np = Kv[s];
    const int ic = (s == 0) ? 3 : Dv[s - 1];
    const int icp = (ic + 31) & ~31;
    const int hhp = (hh + 31) & ~31;
    const int BG = Bv * n, R = BG * np, hd = d / Hn;
    const int sb = s * 73;  // param leaves per stage (sorted pytree keys)
    auto BP = [&](int bl, int j) { return P(sb + bl * 11 + j); };
    // stage leaves: blocks[5x11], nb, ng, pb1, pb2, pw1, pw2, te[12]
    const float* nb_ = P(sb + 55);
    const float* ng_ = P(sb + 56);
    const float* pb1 = P(sb + 57);
    const float* pb2 = P(sb + 58);
    const float* pw1 = P(sb + 59);
    const float* pw2 = P(sb + 60);
    // te: b1,b2,b3,b4,bn1_b,bn1_s,bn2_b,bn2_s,w1,w2,w3,w4
    const float* teb1 = P(sb + 61 + 0);
    const float* teb2 = P(sb + 61 + 1);
    const float* teb3 = P(sb + 61 + 2);
    const float* teb4 = P(sb + 61 + 3);
    const float* bn1b = P(sb + 61 + 4);
    const float* bn1s = P(sb + 61 + 5);
    const float* bn2b = P(sb + 61 + 6);
    const float* bn2s = P(sb + 61 + 7);
    const float* tw1 = P(sb + 61 + 8);
    const float* tw2 = P(sb + 61 + 9);
    const float* tw3 = P(sb + 61 + 10);
    const float* tw4 = P(sb + 61 + 11);

    _Float16* tein = (_Float16*)alloc((size_t)R * icp * 2);
    _Float16* w1t = (_Float16*)alloc((size_t)hh * icp * 2);
    _Float16* w2t = (_Float16*)alloc((size_t)hh * hhp * 2);
    _Float16* w3t = (_Float16*)alloc((size_t)d * d * 2);
    _Float16* w4t = (_Float16*)alloc((size_t)d * d * 2);
    _Float16* pw1t = (_Float16*)alloc((size_t)d * 32 * 2);
    _Float16* pw2t = (_Float16*)alloc((size_t)d * d * 2);
    _Float16 *qkvt[5], *projt[5], *fc1t[5], *fc2t[5];
    for (int bl = 0; bl < 5; ++bl) {
      qkvt[bl] = (_Float16*)alloc((size_t)3 * d * d * 2);
      projt[bl] = (_Float16*)alloc((size_t)d * d * 2);
      fc1t[bl] = (_Float16*)alloc((size_t)4 * d * d * 2);
      fc2t[bl] = (_Float16*)alloc((size_t)4 * d * d * 2);
    }
    float* xbuf = (float*)alloc((size_t)BG * d * 4);
    float* posf = (float*)alloc((size_t)BG * d * 4);
    float* xin = (float*)alloc((size_t)BG * d * 4);
    _Float16* cin = (_Float16*)alloc((size_t)BG * 32 * 2);
    _Float16* posh = (_Float16*)alloc((size_t)BG * d * 2);
    _Float16* yh = (_Float16*)alloc((size_t)BG * d * 2);
    _Float16* qkvh = (_Float16*)alloc((size_t)BG * 3 * d * 2);
    _Float16* oh = (_Float16*)alloc((size_t)BG * d * 2);
    _Float16* mlph = (_Float16*)alloc((size_t)BG * 4 * d * 2);

    // token-embed input (gathered rows, f16, K padded)
    if (s == 0) {
      int tot = R * icp;
      k_cast_pad<<<(tot + 255) / 256, 256, 0, stream>>>(neigh0, tein, R, 3, icp);
    } else {
      int tot = R * dprev;
      k_gather_cast<<<(tot + 255) / 256, 256, 0, stream>>>(xprev, idxs[s - 1],
                                                           tein, R, dprev);
    }
    // weight prep (transpose + pad + cast to f16)
    prep(tw1, ic, hh, w1t, icp);
    prep(tw2, hh, hh, w2t, hhp);
    prep(tw3, d, d, w3t, d);
    prep(tw4, d, d, w4t, d);
    prep(pw1, 3, d, pw1t, 32);
    prep(pw2, d, d, pw2t, d);
    for (int bl = 0; bl < 5; ++bl) {
      prep(BP(bl, 10), d, 3 * d, qkvt[bl], d);   // qkv_w
      prep(BP(bl, 9), d, d, projt[bl], d);       // proj_w
      prep(BP(bl, 1), d, 4 * d, fc1t[bl], d);    // fc1_w
      prep(BP(bl, 3), 4 * d, d, fc2t[bl], 4 * d);// fc2_w
    }
    // fused token embed -> tokens (xbuf)
    size_t lds_te = (size_t)np * hhp * 2 + (size_t)np * hh * 2 +
                    (size_t)np * d * 2 * 2 + (size_t)(np / 16) * d * 4;
    k_token_embed<<<BG, 128, lds_te, stream>>>(tein, w1t, teb1, bn1s, bn1b, w2t,
                                               teb2, w3t, teb3, bn2s, bn2b, w4t,
                                               teb4, xbuf, np, icp, hh, hhp, d);
    // positional MLP: gelu(c@pw1+pb1)@pw2+pb2
    {
      int tot = BG * 32;
      k_cast_pad<<<(tot + 255) / 256, 256, 0, stream>>>(cent[s], cin, BG, 3, 32);
    }
    gemm(cin, 32, pw1t, 32, pb1, nullptr, nullptr, nullptr, nullptr, posh, d, BG,
         d, 32, ACT_GELU);
    gemm(posh, d, pw2t, d, pb2, nullptr, nullptr, nullptr, posf, nullptr, d, BG,
         d, d, ACT_NONE);

    float scale = 1.0f / sqrtf((float)hd);
    float r2 = RAD[s] * RAD[s];
    for (int bl = 0; bl < 5; ++bl) {
      const float* fc1b = BP(bl, 0);
      const float* fc2b = BP(bl, 2);
      const float* ln1b = BP(bl, 4);
      const float* ln1g = BP(bl, 5);
      const float* ln2b = BP(bl, 6);
      const float* ln2g = BP(bl, 7);
      const float* projb = BP(bl, 8);
      int tot = BG * d;
      k_add<<<(tot + 255) / 256, 256, 0, stream>>>(xbuf, posf, xin, tot);
      k_layernorm<<<(BG + 3) / 4, 128, 0, stream>>>(xin, ln1g, ln1b, yh, BG, d);
      gemm(yh, d, qkvt[bl], d, nullptr, nullptr, nullptr, nullptr, nullptr, qkvh,
           3 * d, BG, 3 * d, d, ACT_NONE);
      // LDS: 16*n f32 scores + 16*n f16 probs + hd*n f16 V^T
      size_t lds_at = (size_t)n * 96 + (size_t)hd * n * 2;
      k_attn<<<Bv * Hn * (n / 16), 256, lds_at, stream>>>(qkvh, cent[s], oh, n,
                                                          d, hd, scale, r2);
      gemm(oh, d, projt[bl], d, projb, nullptr, nullptr, xin, xbuf, nullptr, d,
           BG, d, d, ACT_NONE);
      k_layernorm<<<(BG + 3) / 4, 128, 0, stream>>>(xbuf, ln2g, ln2b, yh, BG, d);
      gemm(yh, d, fc1t[bl], d, fc1b, nullptr, nullptr, nullptr, nullptr, mlph,
           4 * d, BG, 4 * d, d, ACT_GELU);
      gemm(mlph, 4 * d, fc2t[bl], 4 * d, fc2b, nullptr, nullptr, xbuf, xbuf,
           nullptr, d, BG, d, 4 * d, ACT_NONE);
    }
    k_ln_out<<<(BG + 3) / 4, 128, 0, stream>>>(xbuf, ng_, nb_, outp + outoff[s],
                                               n, d, BG);
    xprev = xbuf;
    dprev = d;
  }
}
